// MultiheadSelfAttention_78176994722119
// MI455X (gfx1250) — matrix operand order repaired and checked by reading, not compiled, not run on hardware
//
#include <hip/hip_runtime.h>
#include <stdint.h>

#define WIDTH_ 1024
#define HEADS_ 16
#define HDIM_ 64
#define NCTX_ 2048
#define BATCH_ 2
#define ROWS_ (BATCH_ * NCTX_)  // 4096

typedef __bf16 bf16;
typedef __attribute__((ext_vector_type(16))) __bf16 v16bf;
typedef __attribute__((ext_vector_type(8))) float v8f;

__device__ __forceinline__ bf16 cvt_bf16(float f) {
    union { float f; uint32_t u; } in;
    in.f = f;
    uint32_t r = in.u + 0x7FFFu + ((in.u >> 16) & 1u);  // round-to-nearest-even
    union { uint16_t s; bf16 b; } out;
    out.s = (uint16_t)(r >> 16);
    return out.b;
}

union Frag { v16bf v; uint32_t u[8]; };

// A-matrix 16x32 bf16 fragment (ISA 7.12.2): lane L = row (L&15);
// lanes 0-15: K = k0+{0..7} in regs0-3 (2/VGPR), K = k0+{16..23} in regs4-7;
// lanes 16-31: K = k0+{8..15}, k0+{24..31}.
__device__ __forceinline__ v16bf load_a(const bf16* src, int ld, int row0, int k0, int lane) {
    int l16 = lane & 15, half = (lane >> 4) & 1;
    const uint32_t* p = (const uint32_t*)(src + (row0 + l16) * ld);
    Frag f;
    int kb = (k0 + half * 8) >> 1;
#pragma unroll
    for (int r = 0; r < 4; ++r) f.u[r] = p[kb + r];
    kb = (k0 + 16 + half * 8) >> 1;
#pragma unroll
    for (int r = 0; r < 4; ++r) f.u[4 + r] = p[kb + r];
    return f.v;
}

// B-matrix 32x16 bf16 fragment: lane L = column j0+(L&15);
// lanes 0-15 hold K = k0+{0..7} in regs0-3, k0+{16..23} in regs4-7 (2/VGPR); lanes 16-31: K = k0+{8..15}, k0+{24..31}.
// srcT[j * ldt + k] must address element (k, j)  ->  K-contiguous storage.
__device__ __forceinline__ v16bf load_b(const bf16* srcT, int ldt, int j0, int k0, int lane) {
    int l16 = lane & 15, half = (lane >> 4) & 1;
    const uint32_t* p = (const uint32_t*)(srcT + (j0 + l16) * ldt + k0 + half * 8);
    Frag f;
#pragma unroll
    for (int r = 0; r < 8; ++r) f.u[r] = p[(r & 3) + (r >> 2) * 8];
    return f.v;
}

__device__ __forceinline__ v8f wmma_bf16(v16bf a, v16bf b, v8f c) {
    return __builtin_amdgcn_wmma_f32_16x16x32_bf16(false, a, false, b, (short)0, c,
                                                   false, false);
}

// ---- CDNA5 async global->LDS copy (ASYNCcnt-tracked), via inline asm ----
__device__ __forceinline__ void async_load_b128(bf16* lds_dst, const bf16* gsrc) {
    uint32_t loff = (uint32_t)(uintptr_t)lds_dst;       // low 32 bits = LDS byte offset
    uint64_t gaddr = (uint64_t)(uintptr_t)gsrc;
    asm volatile("global_load_async_to_lds_b128 %0, %1, off"
                 :
                 : "v"(loff), "v"(gaddr)
                 : "memory");
}
__device__ __forceinline__ void wait_asynccnt0() {
    asm volatile("s_wait_asynccnt 0" ::: "memory");
}

// ---------------- conversion kernels ----------------
__global__ void cvt_copy_k(const float* __restrict__ src, bf16* __restrict__ dst, int n) {
    int i = blockIdx.x * blockDim.x + threadIdx.x;
    if (i < n) dst[i] = cvt_bf16(src[i]);
}

// src[r][c] (row-major rows x cols, f32) -> dst[c][r] (bf16)
__global__ void cvt_transpose_k(const float* __restrict__ src, bf16* __restrict__ dst,
                                int rows, int cols) {
    int i = blockIdx.x * blockDim.x + threadIdx.x;
    if (i < rows * cols) {
        int r = i / cols, c = i - r * cols;
        dst[c * rows + r] = cvt_bf16(src[i]);
    }
}

// ---------------- QKV projection: [4096,1024] x [1024,3072] ----------------
// One wave computes a 16x64 output tile: load A + all 4 B fragments first,
// then 4 WMMAs, so WMMAs retire against partial load counts (overlap).
__global__ __launch_bounds__(128) void qkv_gemm_k(const bf16* __restrict__ xb,
                                                  const bf16* __restrict__ wT,
                                                  const float* __restrict__ bias,
                                                  bf16* __restrict__ q, bf16* __restrict__ k,
                                                  bf16* __restrict__ vT) {
    int lane = threadIdx.x & 31;
    int wave = threadIdx.x >> 5;
    int j0 = (blockIdx.x * 4 + wave) * 64;
    int row0 = blockIdx.y * 16;
    v8f acc[4] = {};
    for (int kk = 0; kk < WIDTH_; kk += 32) {
        v16bf a = load_a(xb, WIDTH_, row0, kk, lane);
        v16bf b0 = load_b(wT, WIDTH_, j0, kk, lane);
        v16bf b1 = load_b(wT, WIDTH_, j0 + 16, kk, lane);
        v16bf b2 = load_b(wT, WIDTH_, j0 + 32, kk, lane);
        v16bf b3 = load_b(wT, WIDTH_, j0 + 48, kk, lane);
        acc[0] = wmma_bf16(a, b0, acc[0]);
        acc[1] = wmma_bf16(a, b1, acc[1]);
        acc[2] = wmma_bf16(a, b2, acc[2]);
        acc[3] = wmma_bf16(a, b3, acc[3]);
    }
    int l16 = lane & 15, half = (lane >> 4) & 1;
#pragma unroll
    for (int c = 0; c < 4; ++c) {
        int j = j0 + c * 16 + l16;
        float bj = bias[j];
        int hh = j / (3 * HDIM_);
        int r3 = j - hh * (3 * HDIM_);
#pragma unroll
        for (int r = 0; r < 8; ++r) {
            int row = row0 + r + half * 8;
            int bidx = row >> 11;  // / NCTX_
            int n = row & (NCTX_ - 1);
            bf16 val = cvt_bf16(acc[c][r] + bj);
            if (r3 < HDIM_)
                q[((bidx * HEADS_ + hh) * NCTX_ + n) * HDIM_ + r3] = val;
            else if (r3 < 2 * HDIM_)
                k[((bidx * HEADS_ + hh) * NCTX_ + n) * HDIM_ + (r3 - HDIM_)] = val;
            else
                vT[((bidx * HEADS_ + hh) * HDIM_ + (r3 - 2 * HDIM_)) * NCTX_ + n] = val;
        }
    }
}

// ---------------- flash attention ----------------
// One block = 4 waves = 64 queries of one (b, h). K/V tiles (32 keys) are
// double-buffered in LDS via async global->LDS copies and shared by all 4 waves.
__global__ __launch_bounds__(128) void attn_k(const bf16* __restrict__ qg,
                                              const bf16* __restrict__ kg,
                                              const bf16* __restrict__ vTg,
                                              bf16* __restrict__ og) {
    __shared__ __align__(16) bf16 kbuf[2][32 * HDIM_];   // [buf][key][ch]   4KB x2
    __shared__ __align__(16) bf16 vbuf[2][HDIM_ * 32];   // [buf][ch][key]   4KB x2
    __shared__ __align__(16) bf16 pbuf[4][16 * 32];      // per-wave P stage 1KB x4

    int t = threadIdx.x;
    int lane = t & 31;
    int wave = t >> 5;
    int l16 = lane & 15, half = (lane >> 4) & 1;
    int h = blockIdx.y, b = blockIdx.z;
    int row0 = blockIdx.x * 64 + wave * 16;  // query tile for this wave

    const bf16* qh = qg + ((size_t)(b * HEADS_ + h) * NCTX_) * HDIM_;
    const bf16* kh = kg + ((size_t)(b * HEADS_ + h) * NCTX_) * HDIM_;
    const bf16* vh = vTg + ((size_t)(b * HEADS_ + h) * HDIM_) * NCTX_;
    bf16* pw = pbuf[wave];

    v16bf qa0 = load_a(qh, HDIM_, row0, 0, lane);
    v16bf qa1 = load_a(qh, HDIM_, row0, 32, lane);

    float mrun[8], lrun[8];
    v8f accO[4] = {};
#pragma unroll
    for (int r = 0; r < 8; ++r) { mrun[r] = -3.0e38f; lrun[r] = 0.f; }
    const float scale = 0.125f;  // 1/sqrt(64)

    // ---- issue copies for tile 0 into buffer 0 ----
    // K tile: 32 keys x 64 ch row-major = one contiguous 4KB block
    {
        const bf16* ksrc = kh;  // s0 = 0
        async_load_b128(&kbuf[0][t * 8], ksrc + t * 8);
        async_load_b128(&kbuf[0][1024 + t * 8], ksrc + 1024 + t * 8);
#pragma unroll
        for (int rnd = 0; rnd < 2; ++rnd) {
            int cidx = t + rnd * 128;         // 256 16B chunks, 4 per ch-row
            int rowc = cidx >> 2, col = (cidx & 3) * 8;
            async_load_b128(&vbuf[0][rowc * 32 + col], vh + rowc * NCTX_ + col);
        }
    }

    int cur = 0;
    for (int s0 = 0; s0 < NCTX_; s0 += 32, cur ^= 1) {
        wait_asynccnt0();   // our copies into buf[cur] are complete
        __syncthreads();    // all copies visible; all waves done with buf[cur^1]

        // prefetch next tile into the other buffer (overlaps with compute below)
        if (s0 + 32 < NCTX_) {
            int nxt = cur ^ 1;
            const bf16* ksrc = kh + (s0 + 32) * HDIM_;
            async_load_b128(&kbuf[nxt][t * 8], ksrc + t * 8);
            async_load_b128(&kbuf[nxt][1024 + t * 8], ksrc + 1024 + t * 8);
#pragma unroll
            for (int rnd = 0; rnd < 2; ++rnd) {
                int cidx = t + rnd * 128;
                int rowc = cidx >> 2, col = (cidx & 3) * 8;
                async_load_b128(&vbuf[nxt][rowc * 32 + col],
                                vh + rowc * NCTX_ + (s0 + 32) + col);
            }
        }

        const bf16* kb = kbuf[cur];
        const bf16* vb = vbuf[cur];

        // QK^T: fetch all 4 K fragments, then run the 4 WMMAs
        v16bf kf00 = load_b(kb, HDIM_, 0, 0, lane);
        v16bf kf01 = load_b(kb, HDIM_, 0, 32, lane);
        v16bf kf10 = load_b(kb, HDIM_, 16, 0, lane);
        v16bf kf11 = load_b(kb, HDIM_, 16, 32, lane);
        v8f sc0 = {}, sc1 = {};
        sc0 = wmma_bf16(qa0, kf00, sc0);
        sc0 = wmma_bf16(qa1, kf01, sc0);
        sc1 = wmma_bf16(qa0, kf10, sc1);
        sc1 = wmma_bf16(qa1, kf11, sc1);

        float mnew[8], corr[8];
#pragma unroll
        for (int r = 0; r < 8; ++r) {
            sc0[r] *= scale;
            sc1[r] *= scale;
            float tm = fmaxf(sc0[r], sc1[r]);
            tm = fmaxf(tm, __shfl_xor(tm, 1, 32));
            tm = fmaxf(tm, __shfl_xor(tm, 2, 32));
            tm = fmaxf(tm, __shfl_xor(tm, 4, 32));
            tm = fmaxf(tm, __shfl_xor(tm, 8, 32));
            mnew[r] = fmaxf(mrun[r], tm);
            corr[r] = __expf(mrun[r] - mnew[r]);
            mrun[r] = mnew[r];
        }
#pragma unroll
        for (int r = 0; r < 8; ++r) {
            sc0[r] = __expf(sc0[r] - mnew[r]);
            sc1[r] = __expf(sc1[r] - mnew[r]);
            float rs = sc0[r] + sc1[r];
            rs += __shfl_xor(rs, 1, 32);
            rs += __shfl_xor(rs, 2, 32);
            rs += __shfl_xor(rs, 4, 32);
            rs += __shfl_xor(rs, 8, 32);
            lrun[r] = lrun[r] * corr[r] + rs;
        }
#pragma unroll
        for (int c = 0; c < 4; ++c)
#pragma unroll
            for (int r = 0; r < 8; ++r) accO[c][r] *= corr[r];

        // C-layout -> A-layout transpose of P through per-wave LDS tile
#pragma unroll
        for (int r = 0; r < 8; ++r) {
            int m = r + half * 8;
            pw[m * 32 + l16] = cvt_bf16(sc0[r]);
            pw[m * 32 + 16 + l16] = cvt_bf16(sc1[r]);
        }
        v16bf pa = load_a(pw, 32, 0, 0, lane);
        // PV: fetch all 4 V fragments, then run the 4 WMMAs
        v16bf vf0 = load_b(vb, 32, 0, 0, lane);
        v16bf vf1 = load_b(vb, 32, 16, 0, lane);
        v16bf vf2 = load_b(vb, 32, 32, 0, lane);
        v16bf vf3 = load_b(vb, 32, 48, 0, lane);
        accO[0] = wmma_bf16(pa, vf0, accO[0]);
        accO[1] = wmma_bf16(pa, vf1, accO[1]);
        accO[2] = wmma_bf16(pa, vf2, accO[2]);
        accO[3] = wmma_bf16(pa, vf3, accO[3]);

        __syncthreads();  // all waves done reading buf[cur] before it is refilled
    }

#pragma unroll
    for (int c = 0; c < 4; ++c) {
#pragma unroll
        for (int r = 0; r < 8; ++r) {
            int n = row0 + r + half * 8;
            float val = accO[c][r] / lrun[r];
            og[((size_t)(b * NCTX_ + n)) * WIDTH_ + h * HDIM_ + c * 16 + l16] =
                cvt_bf16(val);
        }
    }
}

// ---------------- output projection: [4096,1024] x [1024,1024] + bias -> f32 ----------------
__global__ __launch_bounds__(128) void proj_gemm_k(const bf16* __restrict__ ab,
                                                   const bf16* __restrict__ wT,
                                                   const float* __restrict__ bias,
                                                   float* __restrict__ out) {
    int lane = threadIdx.x & 31;
    int wave = threadIdx.x >> 5;
    int j0 = (blockIdx.x * 4 + wave) * 64;
    int row0 = blockIdx.y * 16;
    v8f acc[4] = {};
    for (int kk = 0; kk < WIDTH_; kk += 32) {
        v16bf a = load_a(ab, WIDTH_, row0, kk, lane);
        v16bf b0 = load_b(wT, WIDTH_, j0, kk, lane);
        v16bf b1 = load_b(wT, WIDTH_, j0 + 16, kk, lane);
        v16bf b2 = load_b(wT, WIDTH_, j0 + 32, kk, lane);
        v16bf b3 = load_b(wT, WIDTH_, j0 + 48, kk, lane);
        acc[0] = wmma_bf16(a, b0, acc[0]);
        acc[1] = wmma_bf16(a, b1, acc[1]);
        acc[2] = wmma_bf16(a, b2, acc[2]);
        acc[3] = wmma_bf16(a, b3, acc[3]);
    }
    int l16 = lane & 15, half = (lane >> 4) & 1;
#pragma unroll
    for (int c = 0; c < 4; ++c) {
        int j = j0 + c * 16 + l16;
        float bj = bias[j];
#pragma unroll
        for (int r = 0; r < 8; ++r) {
            int row = row0 + r + half * 8;
            out[(size_t)row * WIDTH_ + j] = acc[c][r] + bj;
        }
    }
}

extern "C" void kernel_launch(void* const* d_in, const int* in_sizes, int n_in,
                              void* d_out, int out_size, void* d_ws, size_t ws_size,
                              hipStream_t stream) {
    const float* x = (const float*)d_in[0];
    const float* w_qkv = (const float*)d_in[1];
    const float* b_qkv = (const float*)d_in[2];
    const float* w_proj = (const float*)d_in[3];
    const float* b_proj = (const float*)d_in[4];
    float* out = (float*)d_out;

    char* ws = (char*)d_ws;
    // byte offsets (all 256-aligned); total ~48 MB
    bf16* xb     = (bf16*)(ws + 0);          // 4096*1024       bf16  (8 MB)
    bf16* wqkvT  = (bf16*)(ws + 8388608);    // 3072*1024       bf16  (6 MB)
    bf16* wprojT = (bf16*)(ws + 14680064);   // 1024*1024       bf16  (2 MB)
    bf16* qB     = (bf16*)(ws + 16777216);   // [b][h][n][c]          (8 MB)
    bf16* kB     = (bf16*)(ws + 25165824);   // [b][h][n][c]          (8 MB)
    bf16* vTB    = (bf16*)(ws + 33554432);   // [b][h][c][n]          (8 MB)
    bf16* attnB  = (bf16*)(ws + 41943040);   // [b*n][width]          (8 MB)

    cvt_copy_k<<<(ROWS_ * WIDTH_ + 255) / 256, 256, 0, stream>>>(x, xb, ROWS_ * WIDTH_);
    cvt_transpose_k<<<(WIDTH_ * 3 * WIDTH_ + 255) / 256, 256, 0, stream>>>(
        w_qkv, wqkvT, WIDTH_, 3 * WIDTH_);
    cvt_transpose_k<<<(WIDTH_ * WIDTH_ + 255) / 256, 256, 0, stream>>>(
        w_proj, wprojT, WIDTH_, WIDTH_);

    qkv_gemm_k<<<dim3(3 * WIDTH_ / 256, ROWS_ / 16), 128, 0, stream>>>(
        xb, wqkvT, b_qkv, qB, kB, vTB);

    attn_k<<<dim3(NCTX_ / 64, HEADS_, BATCH_), 128, 0, stream>>>(qB, kB, vTB, attnB);

    proj_gemm_k<<<dim3(WIDTH_ / 256, ROWS_ / 16), 128, 0, stream>>>(
        attnB, wprojT, b_proj, out);
}
